// ChronoSynth_29807073035002
// MI455X (gfx1250) — matrix operand order repaired and checked by reading, not compiled, not run on hardware
//
#include <hip/hip_runtime.h>
#include <stdint.h>

// ---------------------------------------------------------------------------
// Types / helpers
// ---------------------------------------------------------------------------
typedef __attribute__((ext_vector_type(16))) __bf16 bf16x16;
typedef __attribute__((ext_vector_type(8)))  float  f32x8;
typedef unsigned int u32x4 __attribute__((ext_vector_type(4)));
typedef int          i32x8 __attribute__((ext_vector_type(8)));
typedef int          i32x4 __attribute__((ext_vector_type(4)));

struct Taps { int n; int dy[9]; int dx[9]; };
struct KIdx { int k[9]; };

__device__ __forceinline__ unsigned short f2bf(float f) {
  union { float f; unsigned u; } v; v.f = f;
  unsigned r = v.u + 0x7fffu + ((v.u >> 16) & 1u);
  return (unsigned short)(r >> 16);
}
__device__ __forceinline__ float bf2f(unsigned short h) {
  union { unsigned u; float f; } v; v.u = ((unsigned)h) << 16; return v.f;
}

struct Q2 { uint4 a, b; };
__device__ __forceinline__ bf16x16 mkfrag(uint4 a, uint4 b) {
  Q2 q; q.a = a; q.b = b;
  return __builtin_bit_cast(bf16x16, q);
}
union CReg { f32x8 v; float f[8]; };

// LDS offset (AS3 ptrtoint) for TDM descriptors
__device__ __forceinline__ unsigned lds_off(const void* p) {
  return (unsigned)(unsigned long long)(__attribute__((address_space(3))) const void*)p;
}

// ---------------------------------------------------------------------------
// Prep: time channel + bf16 packing into padded-NHWC layouts
//   feats : [16(b*4+t)][130][130][96]  bf16 (subnet inputs, Cin=65)
//   occin : [4][130][130][288]         bf16 (fuse input, Cin=260, ch=c*4+t)
// Borders / pad channels pre-zeroed by hipMemsetAsync.
// ---------------------------------------------------------------------------
__global__ __launch_bounds__(256) void prep_feats(
    const float* __restrict__ features, const float* __restrict__ oft,
    unsigned short* __restrict__ feats, unsigned short* __restrict__ occin)
{
  const int total = 16 * 65 * 128 * 128;
  int e = blockIdx.x * 256 + threadIdx.x;
  if (e >= total) return;
  int x = e & 127;
  int y = (e >> 7) & 127;
  int c = (e >> 14) % 65;
  int nt = e / (65 * 16384);
  int b = nt >> 2, t = nt & 3;
  const float ct[4] = {1.25f, -0.25f, 0.0f, 1.0f};
  float v;
  if (c < 64)
    v = features[((((size_t)b * 64 + c) * 4 + t) << 14) + (y << 7) + x];
  else
    v = fabsf(ct[t] - oft[b]);
  unsigned short us = f2bf(v);
  feats[(((size_t)nt * 130 + (y + 1)) * 130 + (x + 1)) * 96 + c] = us;
  occin[(((size_t)b * 130 + (y + 1)) * 130 + (x + 1)) * 288 + (c * 4 + t)] = us;
}

// ---------------------------------------------------------------------------
// Weight prep: pack f32 conv weights into WMMA A-fragment order (bf16).
// Contiguous k order used here (B matches): lane<16: elems0-15 -> K0-15;
// lane>=16: +16.  awgt[((tile*nCb+cb)*ntaps + t)*512 + lane*16 + elem]
// ---------------------------------------------------------------------------
__global__ __launch_bounds__(256) void wprep(
    const float* __restrict__ w, unsigned short* __restrict__ awgt,
    int Cout, int Cin, int ntaps, KIdx kidx, int kElems, int transpose,
    int nCb, int total)
{
  int e = blockIdx.x * 256 + threadIdx.x;
  if (e >= total) return;
  int elem = e & 15;
  int lane = (e >> 4) & 31;
  int rest = e >> 9;
  int t     = rest % ntaps;
  int rest2 = rest / ntaps;
  int cb    = rest2 % nCb;
  int tile  = rest2 / nCb;
  int m  = lane & 15;
  int hi = (lane & 16) ? 16 : 0;
  int k  = elem + hi;
  int cout = tile * 16 + m;
  int cin  = cb * 32 + k;
  unsigned short v = 0;
  if (cout < Cout && cin < Cin) {
    int wi = transpose ? ((cin * Cout + cout) * kElems + kidx.k[t])
                       : ((cout * Cin + cin) * kElems + kidx.k[t]);
    v = f2bf(w[wi]);
  }
  awgt[e] = v;
}

// ---------------------------------------------------------------------------
// Implicit-GEMM conv on WMMA bf16 -> f32, padded-NHWC activations.
// Block = 128 threads = 4 waves; wave = 16 couts x 16 pixels of one row.
// Input tile staged to LDS by the Tensor Data Mover (one 2D descriptor).
// grid: x = (Win/64)*nTiles (strip|tile via wshift), y = row, z = image.
// ---------------------------------------------------------------------------
#define LDSE 19008
__global__ __launch_bounds__(128) void conv_wmma(
    const unsigned short* __restrict__ in, unsigned short* __restrict__ out,
    const unsigned short* __restrict__ awgt, const float* __restrict__ bias,
    int Hin, int Win, int Cs, int Cout, int CsOut, int Hout, int Wout,
    Taps taps, int relu, int upsample, int ry, int rx, int wshift, int nCb)
{
  __shared__ __align__(128) unsigned short lds[LDSE];
  int tid  = threadIdx.x;
  int strip = blockIdx.x & ((1 << wshift) - 1);
  int tile  = blockIdx.x >> wshift;
  int n    = blockIdx.z;
  int y    = blockIdx.y;
  int xb   = strip * 64;

  int dyMin = taps.dy[0], dyMax = taps.dy[0];
  int dxMin = taps.dx[0], dxMax = taps.dx[0];
  for (int t = 1; t < taps.n; t++) {
    dyMin = min(dyMin, taps.dy[t]); dyMax = max(dyMax, taps.dy[t]);
    dxMin = min(dxMin, taps.dx[t]); dxMax = max(dxMax, taps.dx[t]);
  }
  int nrows = dyMax - dyMin + 1;
  int ncols = 64 + dxMax - dxMin;
  int tile0 = ncols * Cs;                       // contiguous elems per tile row
  long rowStride = (long)(Win + 2) * Cs;        // elems per padded image row

  const unsigned short* gsrc = in
      + (size_t)n * (Hin + 2) * rowStride
      + (size_t)(y + 1 + dyMin) * rowStride
      + (size_t)(xb + 1 + dxMin) * Cs;

  if (tid < 32) {  // wave 0 issues the TDM load
    unsigned long long ga = (unsigned long long)(uintptr_t)gsrc;
    unsigned ldso = lds_off(&lds[0]);
    unsigned long long s0 = (unsigned long long)rowStride;  // in elements
    const unsigned td0 = 0x40000000u, td1 = 0x40000000u;    // huge: no OOB
    u32x4 g0;
    g0[0] = 1u;                                             // count=1
    g0[1] = ldso;                                           // lds_addr (bytes)
    g0[2] = (unsigned)(ga & 0xFFFFFFFFull);                 // global_addr lo
    g0[3] = (unsigned)((ga >> 32) & 0x01FFFFFFull) | 0x80000000u; // hi | type=2
    i32x8 g1;
    g1[0] = (int)(1u << 16);                                // data_size=1 (2B)
    g1[1] = (int)((td0 & 0xFFFFu) << 16);                   // tensor_dim0 lo
    g1[2] = (int)((td0 >> 16) | ((td1 & 0xFFFFu) << 16));   // dim0 hi|dim1 lo
    g1[3] = (int)((td1 >> 16) | ((unsigned)tile0 << 16));   // dim1 hi|tile_dim0
    g1[4] = (int)(unsigned)nrows;                           // tile_dim1, dim2=0
    g1[5] = (int)(unsigned)(s0 & 0xFFFFFFFFull);            // dim0_stride lo
    g1[6] = (int)(unsigned)((s0 >> 32) & 0xFFFFull);        // dim0_stride hi
    g1[7] = 0;
    i32x4 g2 = {0, 0, 0, 0}, g3 = {0, 0, 0, 0};
    i32x8 g4 = {0, 0, 0, 0, 0, 0, 0, 0};
    __builtin_amdgcn_tensor_load_to_lds(g0, g1, g2, g3, g4, 0);
    __builtin_amdgcn_s_wait_tensorcnt(0);
  }
  __syncthreads();

  // --- WMMA main loop ---
  int lane  = tid & 31;
  int wave  = tid >> 5;
  int laneN = lane & 15;
  int mrow  = (lane & 16) ? 8 : 0;   // C-matrix M offset for this lane group
  int koff  = (lane & 16) ? 16 : 0;  // B-matrix K offset for this lane group

  CReg c;
#pragma unroll
  for (int i = 0; i < 8; i++) c.f[i] = 0.0f;

  for (int cb = 0; cb < nCb; cb++) {
    const unsigned short* abase = awgt +
        ((size_t)(tile * nCb + cb) * taps.n) * 512 + (size_t)lane * 16;
#pragma unroll
    for (int t = 0; t < 9; t++) {
      if (t >= taps.n) break;  // uniform branch; EXEC stays all-ones
      const uint4* ap = (const uint4*)(abase + (size_t)t * 512);
      uint4 a0 = ap[0], a1 = ap[1];
      int r    = taps.dy[t] - dyMin;
      int cidx = wave * 16 + laneN + taps.dx[t] - dxMin;
      const uint4* bp = (const uint4*)(lds + (size_t)(r * tile0 + cidx * Cs) +
                                       cb * 32 + koff);
      uint4 b0 = bp[0], b1 = bp[1];  // 16 contiguous bf16 (K)
      bf16x16 av = mkfrag(a0, a1);
      bf16x16 bv = mkfrag(b0, b1);
      c.v = __builtin_amdgcn_wmma_f32_16x16x32_bf16(
          false, av, false, bv, (short)0, c.v, false, false);
    }
  }

  // --- epilogue: bias + (leaky)relu + packed b128 store of 8 channels ---
  int ox = xb + wave * 16 + laneN, oy = y;
  if (upsample) { ox = 2 * ox + rx; oy = 2 * oy + ry; }
  union { unsigned short us[8]; uint4 q; } pk;
#pragma unroll
  for (int v = 0; v < 8; v++) {
    int cout = tile * 16 + mrow + v;
    float bb = (cout < Cout) ? bias[cout] : 0.0f;
    float r2 = (cout < Cout) ? (c.f[v] + bb) : 0.0f;
    if (relu) r2 = (r2 < 0.0f) ? 0.2f * r2 : r2;
    pk.us[v] = f2bf(r2);
  }
  size_t obase = (((size_t)n * (Hout + 2) + (oy + 1)) * (Wout + 2) + (ox + 1))
                 * CsOut + tile * 16 + mrow;
  *(uint4*)(out + obase) = pk.q;
}

// ---------------------------------------------------------------------------
// Softmax over 25 channels, in place, bf16, NHWC [16][258][258][32]
// ---------------------------------------------------------------------------
__global__ __launch_bounds__(256) void softmax25_ip(unsigned short* __restrict__ buf)
{
  int idx = blockIdx.x * 256 + threadIdx.x;
  if (idx >= 16 * 65536) return;
  int n = idx >> 16, p = idx & 65535;
  int y = p >> 8, x = p & 255;
  size_t base = (((size_t)n * 258 + (y + 1)) * 258 + (x + 1)) * 32;
  float v[25];
  float m = -1e30f;
#pragma unroll
  for (int c = 0; c < 25; c++) { v[c] = bf2f(buf[base + c]); m = fmaxf(m, v[c]); }
  float s = 0.0f;
#pragma unroll
  for (int c = 0; c < 25; c++) { v[c] = __expf(v[c] - m); s += v[c]; }
  float inv = 1.0f / s;
#pragma unroll
  for (int c = 0; c < 25; c++) buf[base + c] = f2bf(v[c] * inv);
}

// Softmax over 4 inputs: occ3 NHWC [4][258][258][16] bf16 -> occl [4][256][256][4] f32
__global__ __launch_bounds__(256) void softmax4_f32(
    const unsigned short* __restrict__ in, float* __restrict__ outp)
{
  int idx = blockIdx.x * 256 + threadIdx.x;
  if (idx >= 4 * 65536) return;
  int b = idx >> 16, p = idx & 65535;
  int y = p >> 8, x = p & 255;
  size_t base = (((size_t)b * 258 + (y + 1)) * 258 + (x + 1)) * 16;
  float v[4];
  float m = -1e30f;
#pragma unroll
  for (int c = 0; c < 4; c++) { v[c] = bf2f(in[base + c]); m = fmaxf(m, v[c]); }
  float s = 0.0f;
#pragma unroll
  for (int c = 0; c < 4; c++) { v[c] = __expf(v[c] - m); s += v[c]; }
  float inv = 1.0f / s;
  size_t ob = (((size_t)b * 256 + y) * 256 + x) * 4;
#pragma unroll
  for (int c = 0; c < 4; c++) outp[ob + c] = v[c] * inv;
}

// ---------------------------------------------------------------------------
// AdaCoF synthesis: per (b,y,x), 4 inputs x 25 taps x bilinear x 3 channels.
// wq/aq/bq are NHWC [16][258][258][32] bf16; occl [4][256][256][4] f32.
// ---------------------------------------------------------------------------
__global__ __launch_bounds__(256) void synth_kernel(
    const float* __restrict__ frames, const float* __restrict__ occl,
    const unsigned short* __restrict__ wq, const unsigned short* __restrict__ aq,
    const unsigned short* __restrict__ bq, float* __restrict__ outp)
{
  int idx = blockIdx.x * 256 + threadIdx.x;
  if (idx >= 4 * 256 * 256) return;
  int x = idx & 255, y = (idx >> 8) & 255, b = idx >> 16;
  float acc[3] = {0.0f, 0.0f, 0.0f};
  size_t obase = (((size_t)b * 256 + y) * 256 + x) * 4;
  for (int i = 0; i < 4; i++) {
    float oc = occl[obase + i];
    const float* fr = frames + ((size_t)(i * 4 + b)) * 3 * 65536;
    size_t base = (((size_t)(b * 4 + i) * 258 + (y + 1)) * 258 + (x + 1)) * 32;
#pragma unroll 5
    for (int k = 0; k < 25; k++) {
      float wv = bf2f(wq[base + k]) * oc;
      float al = bf2f(aq[base + k]);
      float be = bf2f(bq[base + k]);
      float py = (float)(y + k / 5) + al;   // padded coords (pad=2, Hp=260)
      float px = (float)(x + k % 5) + be;
      int y0 = (int)fminf(fmaxf(floorf(py), 0.0f), 258.0f);
      int x0 = (int)fminf(fmaxf(floorf(px), 0.0f), 258.0f);
      float fy = fminf(fmaxf(py - (float)y0, 0.0f), 1.0f);
      float fx = fminf(fmaxf(px - (float)x0, 0.0f), 1.0f);
      int r0 = min(max(y0 - 2, 0), 255), r1 = min(max(y0 - 1, 0), 255);
      int c0 = min(max(x0 - 2, 0), 255), c1 = min(max(x0 - 1, 0), 255);
      float w00 = (1.0f - fy) * (1.0f - fx) * wv;
      float w01 = (1.0f - fy) * fx * wv;
      float w10 = fy * (1.0f - fx) * wv;
      float w11 = fy * fx * wv;
#pragma unroll
      for (int cch = 0; cch < 3; cch++) {
        const float* f = fr + (size_t)cch * 65536;
        acc[cch] += w00 * f[r0 * 256 + c0] + w01 * f[r0 * 256 + c1] +
                    w10 * f[r1 * 256 + c0] + w11 * f[r1 * 256 + c1];
      }
    }
  }
#pragma unroll
  for (int cch = 0; cch < 3; cch++)
    outp[(((size_t)b * 3 + cch) * 256 + y) * 256 + x] = acc[cch];
}

// ---------------------------------------------------------------------------
// Host side
// ---------------------------------------------------------------------------
static void launch_conv(hipStream_t s,
    const unsigned short* in, unsigned short* out, unsigned short* awgt,
    const float* w, const float* bias,
    int N, int Cin, int Cout, int CsOut, int Hin, int Win,
    const Taps& taps, const KIdx& kidx, int kElems, int transpose, int relu,
    int upsample, int ry, int rx, int Hout, int Wout)
{
  int nTiles = (Cout + 15) / 16;
  int nCb = (Cin + 31) / 32;
  int Cs = nCb * 32;
  int wshift = (Win == 128) ? 1 : 2;   // log2(Win/64)
  int total = nTiles * nCb * taps.n * 512;
  wprep<<<(total + 255) / 256, 256, 0, s>>>(w, awgt, Cout, Cin, taps.n, kidx,
                                            kElems, transpose, nCb, total);
  dim3 g((Win / 64) * nTiles, Hin, N);
  conv_wmma<<<g, 128, 0, s>>>(in, out, awgt, bias, Hin, Win, Cs, Cout, CsOut,
                              Hout, Wout, taps, relu, upsample, ry, rx,
                              wshift, nCb);
}

extern "C" void kernel_launch(void* const* d_in, const int* in_sizes, int n_in,
                              void* d_out, int out_size, void* d_ws, size_t ws_size,
                              hipStream_t stream)
{
  (void)in_sizes; (void)n_in; (void)out_size; (void)ws_size;
  const float* features = (const float*)d_in[0];
  const float* frames   = (const float*)d_in[1];
  const float* oft      = (const float*)d_in[2];
  const int IW = 3, IA = 11, IB = 19, IO = 27;
  const float* fuse_w = (const float*)d_in[35];
  const float* fuse_b = (const float*)d_in[36];
  auto P = [&](int i) { return (const float*)d_in[i]; };

  uint8_t* basep = (uint8_t*)d_ws;
  size_t off = 0;
  auto alloc = [&](size_t bytes) -> uint8_t* {
    uint8_t* p = basep + off;
    off = (off + bytes + 255) & ~((size_t)255);
    return p;
  };
  const size_t P130 = 130ull * 130, P258 = 258ull * 258;
  size_t featsB = 16ull * P130 * 96 * 2;
  size_t occinB = 4ull  * P130 * 288 * 2;
  size_t occfB  = 4ull  * P130 * 96 * 2;
  size_t a1B    = 16ull * P130 * 64 * 2;
  size_t a2B    = 16ull * P130 * 32 * 2;   // >= 4*P130*64*2 (occ)
  size_t atB    = 16ull * P258 * 32 * 2;   // >= 4*P258*64*2 (occ)
  unsigned short* feats = (unsigned short*)alloc(featsB);
  unsigned short* occin = (unsigned short*)alloc(occinB);
  unsigned short* occf  = (unsigned short*)alloc(occfB);
  unsigned short* a1    = (unsigned short*)alloc(a1B);
  unsigned short* a2    = (unsigned short*)alloc(a2B);
  unsigned short* at    = (unsigned short*)alloc(atB);
  unsigned short* wq    = (unsigned short*)alloc(16ull * P258 * 32 * 2);
  unsigned short* aq    = (unsigned short*)alloc(16ull * P258 * 32 * 2);
  unsigned short* bq    = (unsigned short*)alloc(16ull * P258 * 32 * 2);
  unsigned short* occ3  = (unsigned short*)alloc(4ull * P258 * 16 * 2);
  float*          occl  = (float*)alloc(4ull * 65536 * 4 * 4);
  unsigned short* awgt  = (unsigned short*)alloc(256 * 1024);

  // zero borders / pad channels of all conv inputs (~220MB @ 23.3 TB/s ~ 10us)
  (void)hipMemsetAsync(feats, 0, featsB, stream);
  (void)hipMemsetAsync(occin, 0, occinB, stream);
  (void)hipMemsetAsync(occf,  0, occfB,  stream);
  (void)hipMemsetAsync(a1,    0, a1B,    stream);
  (void)hipMemsetAsync(a2,    0, a2B,    stream);
  (void)hipMemsetAsync(at,    0, atB,    stream);

  // tap tables
  Taps t3; t3.n = 9; KIdx k3;
  for (int i = 0; i < 9; i++) { t3.dy[i] = i / 3 - 1; t3.dx[i] = i % 3 - 1; k3.k[i] = i; }
  Taps t1; KIdx k1;
  t1.n = 1; for (int i = 0; i < 9; i++) { t1.dy[i] = 0; t1.dx[i] = 0; k1.k[i] = 0; }
  // ConvT parity sub-kernels: eff[o][i][ky][kx] = w[i][o][2-ky][2-kx]
  Taps tp[4]; KIdx kp[4];
  for (int ry = 0; ry < 2; ry++) for (int rx = 0; rx < 2; rx++) {
    int p = ry * 2 + rx;
    int ny, dys[2], kys[2], nx, dxs[2], kxs[2];
    if (ry == 0) { ny = 1; dys[0] = 0; kys[0] = 1; }
    else { ny = 2; dys[0] = 0; kys[0] = 0; dys[1] = 1; kys[1] = 2; }
    if (rx == 0) { nx = 1; dxs[0] = 0; kxs[0] = 1; }
    else { nx = 2; dxs[0] = 0; kxs[0] = 0; dxs[1] = 1; kxs[1] = 2; }
    int c = 0;
    for (int iy = 0; iy < ny; iy++) for (int ix = 0; ix < nx; ix++) {
      tp[p].dy[c] = dys[iy]; tp[p].dx[c] = dxs[ix];
      kp[p].k[c] = (2 - kys[iy]) * 3 + (2 - kxs[ix]);
      c++;
    }
    tp[p].n = c;
    for (int i = c; i < 9; i++) { tp[p].dy[i] = 0; tp[p].dx[i] = 0; kp[p].k[i] = 0; }
  }

  // 1) features -> padded-NHWC bf16 (+ time channel)
  {
    int total = 16 * 65 * 128 * 128;
    prep_feats<<<(total + 255) / 256, 256, 0, stream>>>(features, oft, feats, occin);
  }
  // 2) fuse 1x1 conv (260 -> 65) + lrelu
  launch_conv(stream, occin, occf, awgt, fuse_w, fuse_b, 4, 260, 65, 96,
              128, 128, t1, k1, 1, 0, 1, 0, 0, 0, 128, 128);
  // 3) weight / alpha / beta subnets (N=16, C2=25, C3=25)
  const int bases[3] = {IW, IA, IB};
  unsigned short* dsts[3] = {wq, aq, bq};
  for (int s = 0; s < 3; s++) {
    int bi = bases[s];
    launch_conv(stream, feats, a1, awgt, P(bi + 0), P(bi + 1), 16, 65, 64, 64,
                128, 128, t3, k3, 9, 0, 1, 0, 0, 0, 128, 128);
    launch_conv(stream, a1, a2, awgt, P(bi + 2), P(bi + 3), 16, 64, 25, 32,
                128, 128, t3, k3, 9, 0, 1, 0, 0, 0, 128, 128);
    for (int p = 0; p < 4; p++)
      launch_conv(stream, a2, at, awgt, P(bi + 4), P(bi + 5), 16, 25, 25, 32,
                  128, 128, tp[p], kp[p], 9, 1, 0, 1, p >> 1, p & 1, 256, 256);
    launch_conv(stream, at, dsts[s], awgt, P(bi + 6), P(bi + 7), 16, 25, 25, 32,
                256, 256, t3, k3, 9, 0, 0, 0, 0, 0, 256, 256);
  }
  // 4) occlusion subnet (N=4, C2=64, C3=4)
  launch_conv(stream, occf, a1, awgt, P(IO + 0), P(IO + 1), 4, 65, 64, 64,
              128, 128, t3, k3, 9, 0, 1, 0, 0, 0, 128, 128);
  launch_conv(stream, a1, a2, awgt, P(IO + 2), P(IO + 3), 4, 64, 64, 64,
              128, 128, t3, k3, 9, 0, 1, 0, 0, 0, 128, 128);
  for (int p = 0; p < 4; p++)
    launch_conv(stream, a2, at, awgt, P(IO + 4), P(IO + 5), 4, 64, 64, 64,
                128, 128, tp[p], kp[p], 9, 1, 0, 1, p >> 1, p & 1, 256, 256);
  launch_conv(stream, at, occ3, awgt, P(IO + 6), P(IO + 7), 4, 64, 4, 16,
              256, 256, t3, k3, 9, 0, 0, 0, 0, 0, 256, 256);
  // 5) softmaxes
  softmax25_ip<<<(16 * 65536 + 255) / 256, 256, 0, stream>>>(wq);
  softmax4_f32<<<(4 * 65536 + 255) / 256, 256, 0, stream>>>(occ3, occl);
  // 6) AdaCoF synthesis -> d_out
  synth_kernel<<<(4 * 65536) / 256, 256, 0, stream>>>(frames, occl, wq, aq, bq,
                                                      (float*)d_out);
}
